// HybridGIN_GAT_Model_48223892800363
// MI455X (gfx1250) — matrix operand order repaired and checked by reading, not compiled, not run on hardware
//
#include <hip/hip_runtime.h>
#include <math.h>

// ---------------------------------------------------------------------------
// HybridGIN_GAT on MI455X (gfx1250, wave32):
//  - Dense GEMMs via v_wmma_f32_16x16x32_bf16 (f32 accumulate), templated
//    epilogue, float4 A loads, bf16 column-major pre-converted weights.
//  - Edge gather/scatter via coalesced VMEM + global_atomic_add_f32
//    (bandwidth-bound part: ~5 GB traffic -> ~200us floor @ 23.3 TB/s).
// ---------------------------------------------------------------------------

#define NNODES 50000
#define NEDGES 800000
#define NGRAPH 256
#define DIM    64

typedef __attribute__((ext_vector_type(16))) __bf16          v16bf;
typedef __attribute__((ext_vector_type(8)))  float           v8f;
typedef __attribute__((ext_vector_type(8)))  unsigned short  u16x8;

union BF16Vec { unsigned short u[16]; u16x8 h[2]; v16bf v; };

__device__ __forceinline__ unsigned short f2bf(float f) {
    unsigned int u = __float_as_uint(f);
    u += 0x7FFFu + ((u >> 16) & 1u);      // round-to-nearest-even
    return (unsigned short)(u >> 16);
}

__device__ __forceinline__ void pack4(unsigned short* dst, float4 a) {
    dst[0] = f2bf(a.x); dst[1] = f2bf(a.y); dst[2] = f2bf(a.z); dst[3] = f2bf(a.w);
}

__device__ __forceinline__ float4 add4(float4 a, float4 b) {
    return make_float4(a.x + b.x, a.y + b.y, a.z + b.z, a.w + b.w);
}

__device__ __forceinline__ void atomicMaxF(float* addr, float val) {
    if (!(val < 0.0f)) atomicMax((int*)addr, __float_as_int(val));
    else               atomicMin((unsigned int*)addr, __float_as_uint(val));
}

// ---------------------------------------------------------------------------
// Weight pre-convert: Wt[col*K + k] = bf16(W[k*m + col])  (column-major bf16)
// ---------------------------------------------------------------------------
__global__ void wconv_kernel(const float* __restrict__ W,
                             unsigned short* __restrict__ Wt, int k, int m)
{
    int idx = blockIdx.x * blockDim.x + threadIdx.x;
    if (idx >= k * m) return;
    int col = idx / k;
    int kk  = idx - col * k;
    Wt[idx] = f2bf(W[(size_t)kk * m + col]);
}

// ---------------------------------------------------------------------------
// WMMA GEMM:  C[n x m] = epi( (A (+A2)) [n x K] @ W [K x m] + bias , bn )
// One wave -> one 16x16 output tile. 128 threads = 4 waves per block.
// EPI: 0 = bias only, 1 = bias+relu, 2 = bias+batchnorm+relu
// ---------------------------------------------------------------------------
template<int K, bool HAS_A2, int EPI>
__global__ void wmma_gemm_kernel(const float* __restrict__ A,
                                 const float* __restrict__ A2,
                                 const unsigned short* __restrict__ Wt, // [m x K] bf16
                                 const float* __restrict__ bias,
                                 const float* __restrict__ bn_g,
                                 const float* __restrict__ bn_b,
                                 const float* __restrict__ bn_m,
                                 const float* __restrict__ bn_v,
                                 float* __restrict__ C,
                                 int n, int m)
{
    const int lane = threadIdx.x & 31;
    const int wave = threadIdx.x >> 5;
    const int mt   = m >> 4;
    const int ct   = blockIdx.y * 4 + wave;        // column tile
    if (ct >= mt) return;
    const int rowBase = blockIdx.x << 4;
    const int colBase = ct << 4;
    const int half = lane >> 4;                    // lane half (0/1)
    const int l15  = lane & 15;
    const int col  = colBase + l15;

    const float4* Arow4  = (const float4*)(A  + (size_t)(rowBase + l15) * K);
    const float4* A2row4 = HAS_A2 ? (const float4*)(A2 + (size_t)(rowBase + l15) * K)
                                  : nullptr;

    v8f acc = {};
#pragma unroll
    for (int k0 = 0; k0 < K; k0 += 32) {
        // ---- A fragment (16-bit A 16x32 layout, ISA 7.12.2):
        //   elems 0..7  -> K = k0 + 8*half + i      (one 32B run)
        //   elems 8..15 -> K = k0 + 16 + 8*half + i (one 32B run)
        const int b0 = (k0 + 8 * half) >> 2;
        const int b1 = (k0 + 16 + 8 * half) >> 2;
        float4 a0 = Arow4[b0], a1 = Arow4[b0 + 1];
        float4 a2 = Arow4[b1], a3 = Arow4[b1 + 1];
        if (HAS_A2) {
            a0 = add4(a0, A2row4[b0]); a1 = add4(a1, A2row4[b0 + 1]);
            a2 = add4(a2, A2row4[b1]); a3 = add4(a3, A2row4[b1 + 1]);
        }
        BF16Vec av;
        pack4(av.u + 0, a0);  pack4(av.u + 4, a1);
        pack4(av.u + 8, a2);  pack4(av.u + 12, a3);

        // ---- B fragment: lane = column, elems 0..7 -> K = k0 + 8*half + i,
        //      elems 8..15 -> K = k0 + 16 + 8*half + i (column-major bf16, 2 x b128)
        const u16x8* Wp = (const u16x8*)(Wt + (size_t)col * K + k0 + 8 * half);
        BF16Vec bv;
        bv.h[0] = Wp[0];
        bv.h[1] = Wp[2];

        acc = __builtin_amdgcn_wmma_f32_16x16x32_bf16(
            false, av.v, false, bv.v, (short)0, acc, false, false);
    }

    // epilogue (lane owns one output column; VGPR r owns row rowBase+8*half+r)
    float bb  = bias[col];
    float bsc = 1.0f, bsh = 0.0f;
    if (EPI == 2) {
        bsc = bn_g[col] * rsqrtf(bn_v[col] + 1e-5f);
        bsh = bn_b[col] - bn_m[col] * bsc;
    }
#pragma unroll
    for (int r = 0; r < 8; ++r) {
        int row = rowBase + half * 8 + r;
        float val = acc[r] + bb;
        if (EPI == 2) val = val * bsc + bsh;
        if (EPI >= 1) val = fmaxf(val, 0.0f);
        C[(size_t)row * m + col] = val;   // half-wave contiguous store
    }
}

// ---------------------------------------------------------------------------
// elementwise / edge kernels
// ---------------------------------------------------------------------------
__global__ void fill_kernel(float* __restrict__ p, long n, float v) {
    long i = (long)blockIdx.x * blockDim.x + threadIdx.x;
    if (i < n) p[i] = v;
}

// GIN: agg[dst][c] += h[src][c], thread per (edge,channel) (coalesced atomics)
__global__ void gin_agg_kernel(const float* __restrict__ h,
                               const int* __restrict__ ei,
                               float* __restrict__ agg, long total)
{
    long idx = (long)blockIdx.x * blockDim.x + threadIdx.x;
    if (idx >= total) return;
    int e = (int)(idx >> 6);          // /64
    int c = (int)(idx & 63);
    int s = ei[e], d = ei[NEDGES + e];
    atomicAdd(&agg[(size_t)d * DIM + c], h[(size_t)s * DIM + c]);
}

// GATv2 attention logits + running segment max. thread per (edge, head).
__global__ void gat_alpha_kernel(const float* __restrict__ xl,
                                 const float* __restrict__ xr,
                                 const int* __restrict__ ei,
                                 const float* __restrict__ att,
                                 float* __restrict__ alpha,
                                 float* __restrict__ amax,
                                 int H, int Wd, int total)
{
    int idx = blockIdx.x * blockDim.x + threadIdx.x;
    if (idx >= total) return;
    int e  = idx / H;
    int hh = idx - e * H;
    int s, d;
    if (e < NEDGES) { s = ei[e]; d = ei[NEDGES + e]; } else { s = d = e - NEDGES; }
    const float4* pl = (const float4*)(xl + (size_t)s * Wd + hh * DIM);
    const float4* pr = (const float4*)(xr + (size_t)d * Wd + hh * DIM);
    const float4* pa = (const float4*)(att + hh * DIM);
    float a = 0.0f;
#pragma unroll 4
    for (int c4 = 0; c4 < DIM / 4; ++c4) {
        float4 l = pl[c4], r = pr[c4], w = pa[c4];
        float v;
        v = l.x + r.x; v = (v > 0.0f) ? v : 0.2f * v; a += w.x * v;
        v = l.y + r.y; v = (v > 0.0f) ? v : 0.2f * v; a += w.y * v;
        v = l.z + r.z; v = (v > 0.0f) ? v : 0.2f * v; a += w.z * v;
        v = l.w + r.w; v = (v > 0.0f) ? v : 0.2f * v; a += w.w * v;
    }
    alpha[idx] = a;
    atomicMaxF(&amax[d * H + hh], a);
}

__global__ void gat_exdenom_kernel(const int* __restrict__ ei,
                                   float* __restrict__ alpha,
                                   const float* __restrict__ amax,
                                   float* __restrict__ denom,
                                   int H, int total)
{
    int idx = blockIdx.x * blockDim.x + threadIdx.x;
    if (idx >= total) return;
    int e = idx / H, hh = idx - e * H;
    int d = (e < NEDGES) ? ei[NEDGES + e] : (e - NEDGES);
    float ex = __expf(alpha[idx] - amax[d * H + hh]);
    alpha[idx] = ex;
    atomicAdd(&denom[d * H + hh], ex);
}

__global__ void gat_coef_kernel(const int* __restrict__ ei,
                                float* __restrict__ alpha,
                                const float* __restrict__ denom,
                                int H, int total)
{
    int idx = blockIdx.x * blockDim.x + threadIdx.x;
    if (idx >= total) return;
    int e = idx / H, hh = idx - e * H;
    int d = (e < NEDGES) ? ei[NEDGES + e] : (e - NEDGES);
    alpha[idx] = alpha[idx] / (denom[d * H + hh] + 1e-16f);
}

// out[dst][c] += xl[src][c] * coef[e,head(c)], thread per (edge,channel)
__global__ void gat_scatter_kernel(const float* __restrict__ xl,
                                   const int* __restrict__ ei,
                                   const float* __restrict__ coef,
                                   float* __restrict__ out,
                                   int H, int Wd, long total)
{
    long idx = (long)blockIdx.x * blockDim.x + threadIdx.x;
    if (idx >= total) return;
    int e = (int)(idx / Wd);
    int c = (int)(idx - (long)e * Wd);
    int hh = c >> 6;
    int s, d;
    if (e < NEDGES) { s = ei[e]; d = ei[NEDGES + e]; } else { s = d = e - NEDGES; }
    atomicAdd(&out[(size_t)d * Wd + c], xl[(size_t)s * Wd + c] * coef[e * H + hh]);
}

__global__ void gat_finalize_kernel(const float* __restrict__ acc,
                                    const float* __restrict__ bias,
                                    float* __restrict__ hout,
                                    int Wd, long total)
{
    long idx = (long)blockIdx.x * blockDim.x + threadIdx.x;
    if (idx >= total) return;
    int c = (int)(idx % Wd);
    hout[idx] = fmaxf(acc[idx] + bias[c], 0.0f);
}

// ---------------------------------------------------------------------------
// pooling + heads
// ---------------------------------------------------------------------------
__global__ void count_kernel(const int* __restrict__ batch, float* __restrict__ counts) {
    int i = blockIdx.x * blockDim.x + threadIdx.x;
    if (i < NNODES) atomicAdd(&counts[batch[i]], 1.0f);
}

__global__ void poolacc_kernel(const float* __restrict__ gin_out,
                               const float* __restrict__ gat_out,
                               const int* __restrict__ batch,
                               float* __restrict__ pooled, long total)
{
    long idx = (long)blockIdx.x * blockDim.x + threadIdx.x;
    if (idx >= total) return;
    int i = (int)(idx >> 7);     // /128
    int c = (int)(idx & 127);
    int g = batch[i];
    float v = (c < DIM) ? gin_out[(size_t)i * DIM + c]
                        : gat_out[(size_t)i * DIM + (c - DIM)];
    atomicAdd(&pooled[g * 2 * DIM + c], v);
}

__global__ void pooldiv_kernel(float* __restrict__ pooled,
                               const float* __restrict__ counts)
{
    int idx = blockIdx.x * blockDim.x + threadIdx.x;
    if (idx >= NGRAPH * 2 * DIM) return;
    pooled[idx] /= fmaxf(counts[idx >> 7], 1.0f);
}

// logits + graph_rep (with lovasz = 0 appended)
__global__ void head_kernel(const float* __restrict__ gr,
                            const float* __restrict__ clf_w,
                            const float* __restrict__ clf_b,
                            float* __restrict__ out)
{
    int g = blockIdx.x * blockDim.x + threadIdx.x;
    if (g >= NGRAPH) return;
    float s = 0.0f;
    for (int c = 0; c < DIM; ++c) s += gr[g * DIM + c] * clf_w[c];
    out[g] = s + clf_b[0];                       // logits [G,1]
    float* grp = out + NGRAPH;                   // graph_rep [G, D+1]
    for (int c = 0; c < DIM; ++c) grp[g * (DIM + 1) + c] = gr[g * DIM + c];
    grp[g * (DIM + 1) + DIM] = 0.0f;             // lovasz stub
}

// ---------------------------------------------------------------------------
// host side
// ---------------------------------------------------------------------------
static inline int cdiv(long a, long b) { return (int)((a + b - 1) / b); }

// epi: 0=bias, 1=bias+relu, 2=bias+bn+relu (only used with A2)
static void launch_gemm(int K, const float* A, const float* A2, const float* W,
                        const float* bias, const float* g, const float* b,
                        const float* mu, const float* vv, float* C,
                        int n, int m, int epi, unsigned short* wbuf,
                        hipStream_t s)
{
    wconv_kernel<<<cdiv((long)K * m, 256), 256, 0, s>>>(W, wbuf, K, m);
    dim3 block(128);
    dim3 grid(n / 16, cdiv(m / 16, 4));
    if (K == 64) {
        if (A2)           wmma_gemm_kernel<64, true, 2><<<grid, block, 0, s>>>(A, A2, wbuf, bias, g, b, mu, vv, C, n, m);
        else if (epi == 1) wmma_gemm_kernel<64, false, 1><<<grid, block, 0, s>>>(A, A2, wbuf, bias, g, b, mu, vv, C, n, m);
        else              wmma_gemm_kernel<64, false, 0><<<grid, block, 0, s>>>(A, A2, wbuf, bias, g, b, mu, vv, C, n, m);
    } else {
        wmma_gemm_kernel<128, false, 0><<<grid, block, 0, s>>>(A, A2, wbuf, bias, g, b, mu, vv, C, n, m);
    }
}

extern "C" void kernel_launch(void* const* d_in, const int* in_sizes, int n_in,
                              void* d_out, int out_size, void* d_ws, size_t ws_size,
                              hipStream_t stream)
{
    const float* x        = (const float*)d_in[0];
    const int*   ei       = (const int*)  d_in[1];
    const int*   batch    = (const int*)  d_in[2];
    const float* gin_w1   = (const float*)d_in[3];
    const float* gin_b1   = (const float*)d_in[4];
    const float* gin_bn_g = (const float*)d_in[5];
    const float* gin_bn_b = (const float*)d_in[6];
    const float* gin_bn_m = (const float*)d_in[7];
    const float* gin_bn_v = (const float*)d_in[8];
    const float* gin_w2   = (const float*)d_in[9];
    const float* gin_b2   = (const float*)d_in[10];
    // gat params: (wl, bl, wr, br, att, bias) x 3 starting at 11
    const float* gatp[18];
    for (int i = 0; i < 18; ++i) gatp[i] = (const float*)d_in[11 + i];
    const float* graph_w = (const float*)d_in[29];
    const float* graph_b = (const float*)d_in[30];
    const float* clf_w   = (const float*)d_in[31];
    const float* clf_b   = (const float*)d_in[32];

    const long N = NNODES, E = NEDGES, Etot = NEDGES + NNODES;

    // workspace carve-up (floats; every chunk a multiple of 4 -> 16B aligned)
    float* ws = (float*)d_ws;
    size_t off = 0;
    auto take = [&](size_t cnt) { float* p = ws + off; off += cnt; return p; };
    float* gat_h   = take(N * 128);
    float* xl      = take(N * 128);
    float* xr      = take(N * 128);
    float* accbuf  = take(N * 128);   // GIN agg (N*64) / GAT out accumulator
    float* zbuf    = take(N * 64);
    float* gin_out = take(N * 64);
    float* alpha   = take(Etot * 2);
    float* amax    = take(N * 2);
    float* denom   = take(N * 2);
    float* pooled  = take(NGRAPH * 128);
    float* gr      = take(NGRAPH * 64);
    float* counts  = take(NGRAPH);
    unsigned short* wbuf = (unsigned short*)take(128 * 128 / 2 + 4); // bf16 weights

    const int TB = 256;
    auto fill = [&](float* p, long n, float v) {
        fill_kernel<<<cdiv(n, TB), TB, 0, stream>>>(p, n, v);
    };

    // ---------------- GIN encoder ----------------
    const float* h = x;
    for (int l = 0; l < 3; ++l) {
        fill(accbuf, N * 64, 0.0f);
        gin_agg_kernel<<<cdiv(E * 64, TB), TB, 0, stream>>>(h, ei, accbuf, E * 64);
        // z = relu(BN((h+agg)@w1 + b1))
        launch_gemm(64, h, accbuf, gin_w1 + l * 64 * 64, gin_b1 + l * 64,
                    gin_bn_g + l * 64, gin_bn_b + l * 64,
                    gin_bn_m + l * 64, gin_bn_v + l * 64,
                    zbuf, NNODES, 64, /*epi=*/2, wbuf, stream);
        // h = relu(z@w2 + b2)
        launch_gemm(64, zbuf, nullptr, gin_w2 + l * 64 * 64, gin_b2 + l * 64,
                    nullptr, nullptr, nullptr, nullptr,
                    gin_out, NNODES, 64, /*epi=*/1, wbuf, stream);
        h = gin_out;
    }

    // ---------------- GATv2 encoder ----------------
    const float* hg = x;
    int Win = 64;
    const int Wouts[3] = {128, 128, 64};
    const int Hs[3]    = {2, 2, 1};
    for (int l = 0; l < 3; ++l) {
        const float* wl_  = gatp[l * 6 + 0];
        const float* bl_  = gatp[l * 6 + 1];
        const float* wr_  = gatp[l * 6 + 2];
        const float* br_  = gatp[l * 6 + 3];
        const float* att_ = gatp[l * 6 + 4];
        const float* bi_  = gatp[l * 6 + 5];
        int Wout = Wouts[l], H = Hs[l];

        launch_gemm(Win, hg, nullptr, wl_, bl_, nullptr, nullptr, nullptr, nullptr,
                    xl, NNODES, Wout, 0, wbuf, stream);
        launch_gemm(Win, hg, nullptr, wr_, br_, nullptr, nullptr, nullptr, nullptr,
                    xr, NNODES, Wout, 0, wbuf, stream);

        fill(amax,  N * H, -INFINITY);
        fill(denom, N * H, 0.0f);
        fill(accbuf, N * Wout, 0.0f);

        int totEH = (int)(Etot * H);
        gat_alpha_kernel<<<cdiv(totEH, TB), TB, 0, stream>>>(
            xl, xr, ei, att_, alpha, amax, H, Wout, totEH);
        gat_exdenom_kernel<<<cdiv(totEH, TB), TB, 0, stream>>>(
            ei, alpha, amax, denom, H, totEH);
        gat_coef_kernel<<<cdiv(totEH, TB), TB, 0, stream>>>(
            ei, alpha, denom, H, totEH);
        gat_scatter_kernel<<<cdiv(Etot * Wout, TB), TB, 0, stream>>>(
            xl, ei, alpha, accbuf, H, Wout, Etot * Wout);
        gat_finalize_kernel<<<cdiv(N * Wout, TB), TB, 0, stream>>>(
            accbuf, bi_, gat_h, Wout, N * Wout);

        hg = gat_h;
        Win = Wout;
    }
    // after layer 2: gat_h holds N x 64 output

    // ---------------- pool + heads ----------------
    fill(counts, NGRAPH, 0.0f);
    fill(pooled, NGRAPH * 128, 0.0f);
    count_kernel<<<cdiv(N, TB), TB, 0, stream>>>(batch, counts);
    poolacc_kernel<<<cdiv(N * 128, TB), TB, 0, stream>>>(
        gin_out, gat_h, batch, pooled, N * 128);
    pooldiv_kernel<<<cdiv(NGRAPH * 128, TB), TB, 0, stream>>>(pooled, counts);

    launch_gemm(128, pooled, nullptr, graph_w, graph_b,
                nullptr, nullptr, nullptr, nullptr,
                gr, NGRAPH, 64, 0, wbuf, stream);

    head_kernel<<<1, NGRAPH, 0, stream>>>(gr, clf_w, clf_b, (float*)d_out);
}
